// MultiHeadedMaskedAttention_24352464570233
// MI455X (gfx1250) — matrix operand order repaired and checked by reading, not compiled, not run on hardware
//
#include <hip/hip_runtime.h>
#include <hip/hip_bf16.h>

// ---------------------------------------------------------------------------
// MultiHeadedMaskedAttention on gfx1250 (MI455X):
//   all four GEMMs via v_wmma_f32_16x16x32_bf16 (fp32 accumulate),
//   flash-attention streaming softmax (no S x S materialization),
//   double-buffered (ping-pong) K loops so VMEM overlaps the WMMA pipe.
// ---------------------------------------------------------------------------

#define DEV static __device__ __forceinline__

constexpr int Dm  = 1024;  // model dim
constexpr int Hn  = 16;    // heads
constexpr int HDs = 64;    // head dim
constexpr int Bn  = 4;     // batch
constexpr int Sn  = 2048;  // seq len

typedef __attribute__((ext_vector_type(16))) __bf16         v16bf;
typedef __attribute__((ext_vector_type(8)))  float          v8f;
typedef __attribute__((ext_vector_type(8)))  unsigned short ush8;

union AB16 {
  v16bf v;
  ush8  h[2];
  __bf16 e[16];
};

DEV unsigned short f2bfu(float f) {
  unsigned u = __builtin_bit_cast(unsigned, f);
  u += 0x7FFFu + ((u >> 16) & 1u);        // round-to-nearest-even
  return (unsigned short)(u >> 16);
}
DEV __bf16 f2bf(float f) {
  unsigned short s = f2bfu(f);
  return __builtin_bit_cast(__bf16, s);
}

DEV v8f wmma_bf16(v16bf a, v16bf b, v8f c) {
  // D(f32 16x16) = A(bf16 16x32) * B(bf16 32x16) + C
  return __builtin_amdgcn_wmma_f32_16x16x32_bf16(
      /*neg_a=*/false, a, /*neg_b=*/false, b,
      /*c_mod=*/(short)0, c, /*reuse_a=*/false, /*reuse_b=*/false);
}

// A-matrix (16x32 bf16) loader from a row-major bf16 matrix.
// Contiguous k order (same as the B loaders): lane (m = lane&15) holds K = lo..lo+15,
// lo = (lane>=16)?16:0 -> two contiguous 16B chunks per lane.
DEV v16bf load_A16x32(const unsigned short* base, int ld) {
  int lane = threadIdx.x & 31;
  int row  = lane & 15;
  int lo   = (lane >> 4) << 4;
  const unsigned short* p = base + (size_t)row * ld + lo;
  AB16 u;
  u.h[0] = *(const ush8*)(p);
  u.h[1] = *(const ush8*)(p + 8);
  return u.v;
}

// B-matrix (32x16 bf16) loader where the K dimension is CONTIGUOUS in memory:
// element B[k][n] = base[n*ld + k].  Contiguous k order (matches A loader): lane (n = lane&15) holds
// K = koff..koff+15, koff = (lane>=16)?16:0 -> two contiguous 16B chunks.
DEV v16bf load_B32x16(const unsigned short* base, int ld) {
  int lane = threadIdx.x & 31;
  int n    = lane & 15;
  int koff = (lane >> 4) << 4;
  const unsigned short* p = base + (size_t)n * ld + koff;
  AB16 u;
  u.h[0] = *(const ush8*)(p);
  u.h[1] = *(const ush8*)(p + 8);
  return u.v;
}

// B-matrix for Q*K^T: B[e][n] = K[kt+n][e0+e], K row-major [S,HD].
// Per lane the key row is fixed and e is contiguous; clamp row for causal tail.
DEV v16bf load_Bk_keys(const unsigned short* kbase, int kt, int e0) {
  int lane = threadIdx.x & 31;
  int n    = lane & 15;
  int koff = (lane >> 4) << 4;
  int row  = kt + n;
  if (row > Sn - 1) row = Sn - 1;   // masked anyway (p == 0)
  const unsigned short* p = kbase + (size_t)row * HDs + e0 + koff;
  AB16 u;
  u.h[0] = *(const ush8*)(p);
  u.h[1] = *(const ush8*)(p + 8);
  return u.v;
}

// --- GEMM helpers for the M=32 tiles --------------------------------------
// Load the two A sub-tiles (rows m..m+15 and m+16..m+31) at K offset kk.
DEV void loadA2(v16bf a[2], const unsigned short* A0, int kk) {
  a[0] = load_A16x32(A0 + kk, Dm);
  a[1] = load_A16x32(A0 + (size_t)16 * Dm + kk, Dm);
}
// Load 4 B tiles (64 output columns); B rows of length Dm, K-contiguous.
DEV void loadB4(v16bf b[4], const unsigned short* Wbase, int kk) {
#pragma unroll
  for (int nt = 0; nt < 4; ++nt)
    b[nt] = load_B32x16(Wbase + (size_t)(nt * 16) * Dm + kk, Dm);
}
// 8 WMMAs: both A sub-tiles against the 4 shared B tiles.
DEV void mma8(v8f acc0[4], v8f acc1[4], const v16bf a[2], const v16bf b[4]) {
#pragma unroll
  for (int nt = 0; nt < 4; ++nt) {
    acc0[nt] = wmma_bf16(a[0], b[nt], acc0[nt]);
    acc1[nt] = wmma_bf16(a[1], b[nt], acc1[nt]);
  }
}

// ---------------------------------------------------------------------------
// Conversion kernels
// ---------------------------------------------------------------------------
__global__ void k_cvt_bf16(const float* __restrict__ src,
                           unsigned short* __restrict__ dst, int n) {
  for (int i = blockIdx.x * blockDim.x + threadIdx.x; i < n;
       i += gridDim.x * blockDim.x)
    dst[i] = f2bfu(src[i]);
}

// W [H, D, HD] fp32 -> Wt [H, HD, D] bf16 (so GEMM-B loads are K-contiguous)
__global__ void k_cvt_wT(const float* __restrict__ W,
                         unsigned short* __restrict__ Wt) {
  int n = Hn * Dm * HDs;
  for (int i = blockIdx.x * blockDim.x + threadIdx.x; i < n;
       i += gridDim.x * blockDim.x) {
    int h = i / (HDs * Dm);
    int j = i - h * (HDs * Dm);
    int e = j / Dm;
    int d = j - e * Dm;
    Wt[i] = f2bfu(W[(size_t)h * Dm * HDs + (size_t)d * HDs + e]);
  }
}

// ---------------------------------------------------------------------------
// QKV projection: per (b,h): [S,D] x [D,HD] -> [S,HD]
// one wave = 32 rows x 64 cols tile; ping-pong double buffer over K.
// z selects Q/K/V; V is stored transposed [HD,S] for the PV GEMM.
// ---------------------------------------------------------------------------
__global__ __launch_bounds__(128) void k_qkv(
    const unsigned short* __restrict__ Xbf,   // [B,S,D]
    const unsigned short* __restrict__ Wt,    // [3,H,HD,D]
    unsigned short* __restrict__ Qb,          // [B,H,S,HD]
    unsigned short* __restrict__ Kb,          // [B,H,S,HD]
    unsigned short* __restrict__ Vt) {        // [B,H,HD,S]
  int wave = threadIdx.x >> 5;
  int lane = threadIdx.x & 31;
  int mt   = blockIdx.x * 4 + wave;           // 0..S/32-1
  int by   = blockIdx.y;                      // b*H + h
  int b    = by >> 4;
  int h    = by & 15;
  int which = blockIdx.z;                     // 0=Q 1=K 2=V

  const unsigned short* A0 = Xbf + ((size_t)b * Sn + (size_t)mt * 32) * Dm;
  const unsigned short* W0 = Wt + ((size_t)which * Hn + h) * (size_t)HDs * Dm;

  v8f acc0[4] = {}, acc1[4] = {};
  v16bf aE[2], bE[4], aO[2], bO[4];
  loadA2(aE, A0, 0);
  loadB4(bE, W0, 0);
  int kk = 0;
#pragma unroll 1
  for (; kk + 64 < Dm; kk += 64) {
    loadA2(aO, A0, kk + 32);          // odd buffer in flight
    loadB4(bO, W0, kk + 32);
    mma8(acc0, acc1, aE, bE);         // compute even
    loadA2(aE, A0, kk + 64);          // even buffer for next iter
    loadB4(bE, W0, kk + 64);
    mma8(acc0, acc1, aO, bO);         // compute odd
  }
  // tail: even buffer holds kk, one odd tile remains at kk+32
  loadA2(aO, A0, kk + 32);
  loadB4(bO, W0, kk + 32);
  mma8(acc0, acc1, aE, bE);
  mma8(acc0, acc1, aO, bO);

  int col = lane & 15;
  int hi8 = (lane >> 4) << 3;
  if (which == 2) {
    unsigned short* op = Vt + (size_t)by * HDs * Sn;
#pragma unroll
    for (int sub = 0; sub < 2; ++sub) {
      v8f* acc = sub ? acc1 : acc0;
#pragma unroll
      for (int nt = 0; nt < 4; ++nt)
#pragma unroll
        for (int r = 0; r < 8; ++r) {
          int srow = mt * 32 + sub * 16 + r + hi8;
          int e    = nt * 16 + col;
          op[(size_t)e * Sn + srow] = f2bfu(acc[nt][r]);
        }
    }
  } else {
    unsigned short* op =
        (which == 0 ? Qb : Kb) + ((size_t)by * Sn + (size_t)mt * 32) * HDs;
#pragma unroll
    for (int sub = 0; sub < 2; ++sub) {
      v8f* acc = sub ? acc1 : acc0;
#pragma unroll
      for (int nt = 0; nt < 4; ++nt)
#pragma unroll
        for (int r = 0; r < 8; ++r)
          op[(size_t)(sub * 16 + r + hi8) * HDs + nt * 16 + col] =
              f2bfu(acc[nt][r]);
    }
  }
}

// ---------------------------------------------------------------------------
// Flash attention: one wave owns a 16-row Q tile, streams 32 keys per step.
// Scores/softmax in fp32; P restaged C-layout -> A-layout through wave-private
// LDS (no block barrier: waves have different causal trip counts; same-wave
// DS ops are in-order per ISA).  V loads are hoisted above the softmax VALU
// so v_exp_f32 work hides their latency.
// ---------------------------------------------------------------------------
__global__ __launch_bounds__(128) void k_attn(
    const unsigned short* __restrict__ Qb,   // [B,H,S,HD]
    const unsigned short* __restrict__ Kb,   // [B,H,S,HD]
    const unsigned short* __restrict__ Vt,   // [B,H,HD,S]
    unsigned short* __restrict__ Ctx) {      // [B,S,D] (head-concat)
  __shared__ float sP[4][16 * 32];

  int wave = threadIdx.x >> 5;
  int lane = threadIdx.x & 31;
  int qt   = blockIdx.x * 4 + wave;          // q tile index, 0..127
  int by   = blockIdx.y;
  int b    = by >> 4;
  int h    = by & 15;

  const unsigned short* Qbase = Qb + ((size_t)by * Sn + (size_t)qt * 16) * HDs;
  const unsigned short* Kbase = Kb + (size_t)by * Sn * HDs;
  const unsigned short* Vtb   = Vt + (size_t)by * HDs * Sn;

  v16bf aq0 = load_A16x32(Qbase, HDs);        // e = 0..31
  v16bf aq1 = load_A16x32(Qbase + 32, HDs);   // e = 32..63

  int col  = lane & 15;
  int hi8  = (lane >> 4) << 3;
  float* myP = &sP[wave][0];

  float mrow[8], lrow[8];
#pragma unroll
  for (int r = 0; r < 8; ++r) { mrow[r] = -3.0e38f; lrow[r] = 0.0f; }
  v8f o[4] = {};

  for (int kb = 0; kb <= qt * 16; kb += 32) {
    // per-lane prefetch of the next key tile's rows
    {
      int nrow = kb + 32 + lane;
      if (nrow > Sn - 1) nrow = Sn - 1;
      __builtin_prefetch(Kbase + (size_t)nrow * HDs, 0, 1);
    }
    v8f sacc[2];
#pragma unroll
    for (int t = 0; t < 2; ++t) {
      int kt = kb + t * 16;
      v16bf bk0 = load_Bk_keys(Kbase, kt, 0);
      v16bf bk1 = load_Bk_keys(Kbase, kt, 32);
      v8f z = {};
      z = wmma_bf16(aq0, bk0, z);
      z = wmma_bf16(aq1, bk1, z);
      sacc[t] = z;
    }
    // V tiles in flight while the softmax VALU below executes
    v16bf bv[4];
#pragma unroll
    for (int nt = 0; nt < 4; ++nt)
      bv[nt] = load_B32x16(Vtb + (size_t)(nt * 16) * Sn + kb, Sn);

    // scale + causal mask + online softmax (rows live per 16-lane half)
#pragma unroll
    for (int r = 0; r < 8; ++r) {
      int rg = qt * 16 + r + hi8;
      float s0 = sacc[0][r] * 0.125f;    // 1/sqrt(64)
      float s1 = sacc[1][r] * 0.125f;
      if (kb + col > rg)      s0 = -1.0e30f;
      if (kb + 16 + col > rg) s1 = -1.0e30f;
      float tmax = fmaxf(s0, s1);
#pragma unroll
      for (int dlt = 1; dlt < 16; dlt <<= 1)
        tmax = fmaxf(tmax, __shfl_xor(tmax, dlt, 32));
      float mn = fmaxf(mrow[r], tmax);
      float p0 = __expf(s0 - mn);
      float p1 = __expf(s1 - mn);
      float rs = p0 + p1;
#pragma unroll
      for (int dlt = 1; dlt < 16; dlt <<= 1)
        rs += __shfl_xor(rs, dlt, 32);
      float sc = __expf(mrow[r] - mn);
      lrow[r] = lrow[r] * sc + rs;
      mrow[r] = mn;
#pragma unroll
      for (int nt = 0; nt < 4; ++nt) o[nt][r] *= sc;
      myP[(r + hi8) * 32 + col]      = p0;
      myP[(r + hi8) * 32 + 16 + col] = p1;
    }
    // P (16x32) C-layout -> A-layout via LDS
    AB16 pa;
    {
      int arow = col;          // A row = lane&15
      int lo   = hi8 * 2;      // 0 or 16
#pragma unroll
      for (int j = 0; j < 8; ++j) {
        pa.e[j]     = f2bf(myP[arow * 32 + lo + j]);
        pa.e[8 + j] = f2bf(myP[arow * 32 + 8 + lo + j]);
      }
    }
    // O += P * V
#pragma unroll
    for (int nt = 0; nt < 4; ++nt) o[nt] = wmma_bf16(pa.v, bv[nt], o[nt]);
  }

  // normalize and store head-concat context (bf16)
  float inv[8];
#pragma unroll
  for (int r = 0; r < 8; ++r) inv[r] = 1.0f / lrow[r];
#pragma unroll
  for (int nt = 0; nt < 4; ++nt)
#pragma unroll
    for (int r = 0; r < 8; ++r) {
      size_t row = (size_t)b * Sn + qt * 16 + r + hi8;
      Ctx[row * Dm + h * HDs + nt * 16 + col] = f2bfu(o[nt][r] * inv[r]);
    }
}

// ---------------------------------------------------------------------------
// Output projection: out = Ctx @ Wo^T + bo.
// B[k=j][n=i] = Wo[i*D + j] -> K-contiguous with Wo stored row-major bf16.
// One wave = 32 rows x 64 cols; ping-pong double buffer over K.
// ---------------------------------------------------------------------------
__global__ __launch_bounds__(128) void k_oproj(
    const unsigned short* __restrict__ Ctx,   // [B*S, D] bf16
    const unsigned short* __restrict__ Wob,   // [D, D]   bf16
    const float* __restrict__ bo,             // [D]
    float* __restrict__ out) {                // [B*S, D]
  int wave = threadIdx.x >> 5;
  int lane = threadIdx.x & 31;
  int mt   = blockIdx.x * 4 + wave;           // 0..B*S/32-1
  int ng   = blockIdx.y;                      // 64-col group, 0..15

  const unsigned short* A0 = Ctx + (size_t)mt * 32 * Dm;
  const unsigned short* W0 = Wob + (size_t)(ng * 64) * Dm;

  v8f acc0[4] = {}, acc1[4] = {};
  v16bf aE[2], bE[4], aO[2], bO[4];
  loadA2(aE, A0, 0);
  loadB4(bE, W0, 0);
  int kk = 0;
#pragma unroll 1
  for (; kk + 64 < Dm; kk += 64) {
    loadA2(aO, A0, kk + 32);
    loadB4(bO, W0, kk + 32);
    mma8(acc0, acc1, aE, bE);
    loadA2(aE, A0, kk + 64);
    loadB4(bE, W0, kk + 64);
    mma8(acc0, acc1, aO, bO);
  }
  loadA2(aO, A0, kk + 32);
  loadB4(bO, W0, kk + 32);
  mma8(acc0, acc1, aE, bE);
  mma8(acc0, acc1, aO, bO);

  int col = lane & 15;
  int hi8 = (lane >> 4) << 3;
#pragma unroll
  for (int sub = 0; sub < 2; ++sub) {
    v8f* acc = sub ? acc1 : acc0;
#pragma unroll
    for (int nt = 0; nt < 4; ++nt)
#pragma unroll
      for (int r = 0; r < 8; ++r) {
        int n = ng * 64 + nt * 16 + col;
        out[((size_t)mt * 32 + sub * 16 + r + hi8) * Dm + n] =
            acc[nt][r] + bo[n];
      }
  }
}

// ---------------------------------------------------------------------------
// Host launch
// ---------------------------------------------------------------------------
extern "C" void kernel_launch(void* const* d_in, const int* in_sizes, int n_in,
                              void* d_out, int out_size, void* d_ws,
                              size_t ws_size, hipStream_t stream) {
  const float* emb = (const float*)d_in[0];  // [B,S,D]
  const float* Wq  = (const float*)d_in[1];  // [H,D,HD]
  const float* Wk  = (const float*)d_in[2];
  const float* Wv  = (const float*)d_in[3];
  const float* Wo  = (const float*)d_in[4];  // [D,D]
  const float* bo  = (const float*)d_in[5];  // [D]
  float* out = (float*)d_out;

  // workspace layout (bf16 elements); total ~92.3 MB
  constexpr size_t N_X   = (size_t)Bn * Sn * Dm;        // 8388608
  constexpr size_t N_W1  = (size_t)Hn * HDs * Dm;       // 1048576
  constexpr size_t N_QKV = (size_t)Bn * Hn * Sn * HDs;  // 8388608
  constexpr size_t OFF_XBF = 0;
  constexpr size_t OFF_WT  = OFF_XBF + N_X;             // [3,H,HD,D]
  constexpr size_t OFF_WO  = OFF_WT + 3 * N_W1;
  constexpr size_t OFF_Q   = OFF_WO + (size_t)Dm * Dm;
  constexpr size_t OFF_K   = OFF_Q + N_QKV;
  constexpr size_t OFF_VT  = OFF_K + N_QKV;
  constexpr size_t OFF_CTX = OFF_VT + N_QKV + 64;       // +pad for causal tail reads

  unsigned short* ws  = (unsigned short*)d_ws;
  unsigned short* xbf = ws + OFF_XBF;
  unsigned short* wt  = ws + OFF_WT;
  unsigned short* wob = ws + OFF_WO;
  unsigned short* qb  = ws + OFF_Q;
  unsigned short* kb  = ws + OFF_K;
  unsigned short* vt  = ws + OFF_VT;
  unsigned short* ctx = ws + OFF_CTX;

  k_cvt_bf16<<<4096, 256, 0, stream>>>(emb, xbf, (int)N_X);
  k_cvt_wT<<<2048, 256, 0, stream>>>(Wq, wt + 0 * N_W1);
  k_cvt_wT<<<2048, 256, 0, stream>>>(Wk, wt + 1 * N_W1);
  k_cvt_wT<<<2048, 256, 0, stream>>>(Wv, wt + 2 * N_W1);
  k_cvt_bf16<<<2048, 256, 0, stream>>>(Wo, wob, Dm * Dm);

  k_qkv<<<dim3(Sn / 32 / 4, Bn * Hn, 3), 128, 0, stream>>>(xbf, wt, qb, kb, vt);
  k_attn<<<dim3(Sn / 16 / 4, Bn * Hn), 128, 0, stream>>>(qb, kb, vt, ctx);
  k_oproj<<<dim3(Bn * Sn / 32 / 4, Dm / 64), 128, 0, stream>>>(ctx, wob, bo, out);
}
